// Model_DIHGNN_64149631533096
// MI455X (gfx1250) — matrix operand order repaired and checked by reading, not compiled, not run on hardware
//
#include <hip/hip_runtime.h>
#include <hip/hip_bf16.h>
#include <math.h>

typedef __attribute__((ext_vector_type(16))) __bf16 v16bf;
typedef __attribute__((ext_vector_type(8)))  float  v8f;
typedef int v4i __attribute__((vector_size(16)));   // pointee type expected by async-LDS builtins

#define NNODE 4096
#define DHID  128
#define DIN   148
#define DOUT  148
#define DPAD  160      // DOUT padded to multiple of 32 for WMMA K-loop
#define ITERS 64       // Neumann iterations for (I - alpha*A)^-1
#define BSTRIDE 40     // LDS B-panel row stride (32 + 8 pad, bf16) -> conflict-spread, 16B aligned

// ---- gfx1250 async global->LDS path (guarded: falls back to plain LDS stores) ----
#if defined(__has_builtin)
#if __has_builtin(__builtin_amdgcn_global_load_async_to_lds_b128) && \
    __has_builtin(__builtin_amdgcn_s_wait_asynccnt)
#define HAVE_ASYNC 1
#endif
#endif
#ifndef HAVE_ASYNC
#define HAVE_ASYNC 0
#endif

#if HAVE_ASYNC
#define AS_GLOBAL __attribute__((address_space(1)))
#define AS_LDS    __attribute__((address_space(3)))
__device__ __forceinline__ void async_b128(const __bf16* g, __bf16* l) {
    __builtin_amdgcn_global_load_async_to_lds_b128((AS_GLOBAL v4i*)g, (AS_LDS v4i*)l, 0, 0);
}
#endif

// ---------------- workspace layout (bytes, all 1KB-aligned) ----------------
static constexpr size_t OFF_MASK = 0;                                   // 4096*128 u32 = 2MB adjacency bitmask
static constexpr size_t OFF_RS   = OFF_MASK + (size_t)NNODE * 128 * 4;  // row sums   u32[4096]
static constexpr size_t OFF_CS   = OFF_RS   + (size_t)NNODE * 4;        // col sums   u32[4096]
static constexpr size_t OFF_LIS  = OFF_CS   + (size_t)NNODE * 4;        // f32[4096]
static constexpr size_t OFF_RIS  = OFF_LIS  + (size_t)NNODE * 4;        // f32[4096]
static constexpr size_t OFF_SCAL = OFF_RIS  + (size_t)NNODE * 4;        // alpha, 1-alpha
static constexpr size_t OFF_X0   = OFF_SCAL + 1024;                     // f32 [N][128]  (b of iteration)
static constexpr size_t OFF_X    = OFF_X0   + (size_t)NNODE * DHID * 4; // f32 [N][128]  current iterate
static constexpr size_t OFF_Y    = OFF_X    + (size_t)NNODE * DHID * 4; // f32 [N][128]  GEMM output
static constexpr size_t OFF_XT   = OFF_Y    + (size_t)NNODE * DHID * 4; // bf16 [128][N] transposed iterate
static constexpr size_t OFF_XD   = OFF_XT   + (size_t)DHID * NNODE * 2; // f32 [N][148]
static constexpr size_t OFF_XDP  = OFF_XD   + (size_t)NNODE * DOUT * 4; // bf16 [N][160] padded
static constexpr size_t OFF_POOL = OFF_XDP  + (size_t)NNODE * DPAD * 2; // u32[148] max-pool bits
static constexpr size_t OFF_LOSS = OFF_POOL + 1024;                     // f32 accumulator
static constexpr size_t OFF_AA   = OFF_LOSS + 1024;                     // bf16 [N][N] = alpha*A_norm, 32MB

// ---------------- small kernels ----------------
__global__ void k_scalars(const float* a, float* scal) {
    float al = 0.5f + 0.5f * tanhf(2.0f * a[0]);
    scal[0] = al; scal[1] = 1.0f - al;
}

__global__ void k_init(unsigned* mask, unsigned* pooled, float* loss) {
    int t = blockIdx.x * blockDim.x + threadIdx.x;
    if (t < NNODE * 128) mask[t] = 0u;
    if (t < DOUT) pooled[t] = 0u;
    if (t == 0) *loss = 0.0f;
}

__global__ void k_scatter(const int* ei, int E, unsigned* mask) {
    int e = blockIdx.x * blockDim.x + threadIdx.x;
    if (e >= E) return;
    int r = ei[e]     & (NNODE - 1);
    int c = ei[E + e] & (NNODE - 1);
    atomicOr(&mask[r * 128 + (c >> 5)], 1u << (c & 31));
}

__global__ void k_rowsum(const unsigned* mask, unsigned* rs) {
    int r = blockIdx.x * blockDim.x + threadIdx.x;
    if (r >= NNODE) return;
    unsigned s = 0;
    for (int w = 0; w < 128; ++w) s += __popc(mask[(size_t)r * 128 + w]);
    rs[r] = s;
}

__global__ void k_colsum(const unsigned* mask, unsigned* cs) {
    int c = blockIdx.x * blockDim.x + threadIdx.x;
    if (c >= NNODE) return;
    unsigned s = 0, w = c >> 5, b = c & 31;
    for (int r = 0; r < NNODE; ++r) s += (mask[(size_t)r * 128 + w] >> b) & 1u;
    cs[c] = s;
}

__global__ void k_norm(const unsigned* cs, const unsigned* rs, float* Lis, float* Ris) {
    int i = blockIdx.x * blockDim.x + threadIdx.x;
    if (i >= NNODE) return;
    Lis[i] = cs[i] ? rsqrtf((float)cs[i]) : 0.0f;
    Ris[i] = rs[i] ? rsqrtf((float)rs[i]) : 0.0f;
}

// materialize Aa[i][j] = alpha * L_is[i] * adj[i][j] * R_is[j] as bf16 (alpha folded in)
__global__ void k_fillA(const unsigned* mask, const float* Lis, const float* Ris,
                        const float* scal, __bf16* Aa) {
    int w = blockIdx.x * blockDim.x + threadIdx.x;   // one thread per 32-bit mask word
    if (w >= NNODE * 128) return;
    int i = w >> 7, j0 = (w & 127) << 5;
    unsigned m = mask[w];
    float s = scal[0] * Lis[i];
    __bf16* out = Aa + (size_t)i * NNODE + j0;
    for (int b = 0; b < 32; ++b)
        out[b] = (__bf16)(((m >> b) & 1u) ? s * Ris[j0 + b] : 0.0f);
}

// x0 = node_feat @ W1^T + b1 ; also seed X = x0 and Xt = bf16(x0^T)
__global__ void k_linear1(const float* nf, const float* W1, const float* b1,
                          float* x0, float* X, __bf16* Xt) {
    int t = blockIdx.x * blockDim.x + threadIdx.x;
    if (t >= NNODE * DHID) return;
    int n = t / DHID, d = t % DHID;
    const float* xr = nf + (size_t)n * DIN;
    const float* wr = W1 + (size_t)d * DIN;
    float acc = b1[d];
    for (int k = 0; k < DIN; ++k) acc += xr[k] * wr[k];
    x0[t] = acc; X[t] = acc;
    Xt[(size_t)d * NNODE + n] = (__bf16)acc;
}

// ---------------- WMMA GEMM: Y[N][128] = Aa[N][N] (bf16) @ X (via Xt[128][N] bf16), f32 accum
// Block = 4 waves, 64 output rows, full 128-col N. B panel (32x128) staged in LDS per k-step,
// shared by all 4 waves; each wave keeps 8 accumulators and reuses its A tile across 8 WMMAs.
__global__ __launch_bounds__(128) void k_gemm(const __bf16* __restrict__ Aa,
                                              const __bf16* __restrict__ Xt,
                                              float* __restrict__ Y) {
    __shared__ __bf16 Bs[DHID * BSTRIDE];            // 10 KB panel, feat-major rows of 32 (+8 pad)
    const int tid  = threadIdx.x;
    const int lane = tid & 31;
    const int wave = tid >> 5;
    const int l15 = lane & 15, h = lane >> 4;
    const int rowBase = blockIdx.x * 64 + wave * 16;
    const __bf16* pa = Aa + (size_t)(rowBase + l15) * NNODE + h * 8;   // A 16x32 layout
    const __bf16* gB = Xt + (size_t)tid * NNODE;                       // thread t stages feature row t
    __bf16* sB = &Bs[tid * BSTRIDE];
    v8f acc[8] = {};
    union U { v16bf v; uint4 q[2]; };
    for (int k = 0; k < NNODE; k += 32) {
        __syncthreads();                             // previous panel fully consumed
#if HAVE_ASYNC
        async_b128(gB + k,      sB);                 // global_load_async_to_lds_b128 x4 (64B/thread)
        async_b128(gB + k + 8,  sB + 8);
        async_b128(gB + k + 16, sB + 16);
        async_b128(gB + k + 24, sB + 24);
        __builtin_amdgcn_s_wait_asynccnt(0);
#else
        const uint4* g = (const uint4*)(gB + k);
        uint4* s = (uint4*)sB;
        s[0] = g[0]; s[1] = g[1]; s[2] = g[2]; s[3] = g[3];
#endif
        __syncthreads();                             // panel visible block-wide
        U a;
        a.q[0] = *(const uint4*)(pa + k);            // K = k + h*8   .. +7
        a.q[1] = *(const uint4*)(pa + k + 16);       // K = k+16+h*8  .. +7
        __builtin_prefetch(pa + k + 256, 0, 3);      // near-scope prefetch of A stream
#pragma unroll
        for (int n = 0; n < 8; ++n) {                // reuse A tile across all 8 N-tiles
            U b;
            const uint4* bs = (const uint4*)&Bs[(n * 16 + l15) * BSTRIDE + h * 8];
            b.q[0] = bs[0]; b.q[1] = bs[2];
            acc[n] = __builtin_amdgcn_wmma_f32_16x16x32_bf16(false, a.v, false, b.v,
                                                             (short)0, acc[n], false, false);
        }
    }
#pragma unroll
    for (int n = 0; n < 8; ++n) {
        float* yo = Y + (size_t)(rowBase + h * 8) * DHID + n * 16 + l15;
#pragma unroll
        for (int r = 0; r < 8; ++r) yo[(size_t)r * DHID] = acc[n][r];
    }
}

// X = x0 + Y  (alpha already folded into Aa); refresh Xt
__global__ void k_update(const float* x0, const float* Y, float* X, __bf16* Xt) {
    int t = blockIdx.x * blockDim.x + threadIdx.x;
    if (t >= NNODE * DHID) return;
    float v = x0[t] + Y[t];
    X[t] = v;
    int n = t / DHID, d = t % DHID;
    Xt[(size_t)d * NNODE + n] = (__bf16)v;
}

// xd = X @ Wd^T + bd
__global__ void k_lineard(const float* X, const float* Wd, const float* bd, float* xd) {
    int t = blockIdx.x * blockDim.x + threadIdx.x;
    if (t >= NNODE * DOUT) return;
    int n = t / DOUT, j = t % DOUT;
    const float* xr = X  + (size_t)n * DHID;
    const float* wr = Wd + (size_t)j * DHID;
    float acc = bd[j];
    for (int k = 0; k < DHID; ++k) acc += xr[k] * wr[k];
    xd[t] = acc;
}

__global__ void k_prepxdp(const float* xd, __bf16* xdp) {
    int t = blockIdx.x * blockDim.x + threadIdx.x;
    if (t >= NNODE * DPAD) return;
    int n = t / DPAD, j = t % DPAD;
    xdp[t] = (__bf16)((j < DOUT) ? xd[(size_t)n * DOUT + j] : 0.0f);
}

// similarity loss: sum over i!=j of relu(xd_i . xd_j), 16x16 WMMA tiles,
// A tile reused across 4 j-tiles per wave.
__global__ __launch_bounds__(32) void k_simloss(const __bf16* __restrict__ xdp, float* loss) {
    const int lane = threadIdx.x;
    const int l15 = lane & 15, h = lane >> 4;
    const int iBase = blockIdx.x * 16;
    const int jBase0 = blockIdx.y * 64;              // 4 j-tiles per wave
    const __bf16* pa = xdp + (size_t)(iBase + l15) * DPAD + h * 8;
    v8f acc[4] = {};
    union U { v16bf v; uint4 q[2]; };
#pragma unroll
    for (int k = 0; k < DPAD; k += 32) {
        U a;
        a.q[0] = *(const uint4*)(pa + k);
        a.q[1] = *(const uint4*)(pa + k + 16);
#pragma unroll
        for (int n = 0; n < 4; ++n) {
            const __bf16* pb = xdp + (size_t)(jBase0 + n * 16 + l15) * DPAD + h * 8;
            U b;
            b.q[0] = *(const uint4*)(pb + k);
            b.q[1] = *(const uint4*)(pb + k + 16);
            acc[n] = __builtin_amdgcn_wmma_f32_16x16x32_bf16(false, a.v, false, b.v,
                                                             (short)0, acc[n], false, false);
        }
    }
    float s = 0.0f;
#pragma unroll
    for (int n = 0; n < 4; ++n) {
        int gj = jBase0 + n * 16 + l15;
#pragma unroll
        for (int r = 0; r < 8; ++r) {
            int gi = iBase + h * 8 + r;
            float v = acc[n][r];
            if (gi != gj && v > 0.0f) s += v;
        }
    }
    for (int off = 16; off > 0; off >>= 1) s += __shfl_xor(s, off, 32);
    if (lane == 0) atomicAdd(loss, s);
}

// global max pool of relu((1-alpha)*xd) via monotone uint atomicMax (values >= 0)
__global__ void k_pool(const float* xd, const float* scal, unsigned* pooled) {
    int t = blockIdx.x * blockDim.x + threadIdx.x;
    if (t >= NNODE * DOUT) return;
    float v = fmaxf(0.0f, scal[1] * xd[t]);
    atomicMax(&pooled[t % DOUT], __float_as_uint(v));
}

__global__ void k_final(const unsigned* pooled, const float* loss,
                        const float* W2, const float* b2, float* out) {
    if (threadIdx.x == 0 && blockIdx.x == 0) {
        for (int k2 = 0; k2 < 2; ++k2) {
            float acc = b2[k2];
            for (int j = 0; j < DOUT; ++j)
                acc += __uint_as_float(pooled[j]) * W2[k2 * DOUT + j];
            out[k2] = acc;
        }
        out[2] = loss[0] * (1.0f / sqrtf((float)DHID)) / ((float)NNODE * (float)(NNODE - 1));
    }
}

// ---------------- launch ----------------
extern "C" void kernel_launch(void* const* d_in, const int* in_sizes, int n_in,
                              void* d_out, int out_size, void* d_ws, size_t ws_size,
                              hipStream_t stream) {
    const float* nf  = (const float*)d_in[0];
    const int*   ei  = (const int*)d_in[1];
    const float* W1  = (const float*)d_in[3];
    const float* b1  = (const float*)d_in[4];
    const float* ap  = (const float*)d_in[5];
    const float* Wd  = (const float*)d_in[6];
    const float* bd  = (const float*)d_in[7];
    const float* W2  = (const float*)d_in[8];
    const float* b2  = (const float*)d_in[9];
    float* outp = (float*)d_out;

    char* ws = (char*)d_ws;
    unsigned* mask   = (unsigned*)(ws + OFF_MASK);
    unsigned* rs     = (unsigned*)(ws + OFF_RS);
    unsigned* cs     = (unsigned*)(ws + OFF_CS);
    float*    Lis    = (float*)(ws + OFF_LIS);
    float*    Ris    = (float*)(ws + OFF_RIS);
    float*    scal   = (float*)(ws + OFF_SCAL);
    float*    x0     = (float*)(ws + OFF_X0);
    float*    X      = (float*)(ws + OFF_X);
    float*    Y      = (float*)(ws + OFF_Y);
    __bf16*   Xt     = (__bf16*)(ws + OFF_XT);
    float*    xd     = (float*)(ws + OFF_XD);
    __bf16*   xdp    = (__bf16*)(ws + OFF_XDP);
    unsigned* pooled = (unsigned*)(ws + OFF_POOL);
    float*    lossa  = (float*)(ws + OFF_LOSS);
    __bf16*   Aa     = (__bf16*)(ws + OFF_AA);

    const int E = in_sizes[1] / 2;

    k_scalars<<<1, 1, 0, stream>>>(ap, scal);
    k_init<<<(NNODE * 128 + 255) / 256, 256, 0, stream>>>(mask, pooled, lossa);
    k_scatter<<<(E + 255) / 256, 256, 0, stream>>>(ei, E, mask);
    k_rowsum<<<NNODE / 256, 256, 0, stream>>>(mask, rs);
    k_colsum<<<NNODE / 256, 256, 0, stream>>>(mask, cs);
    k_norm<<<NNODE / 256, 256, 0, stream>>>(cs, rs, Lis, Ris);
    k_fillA<<<(NNODE * 128) / 128, 128, 0, stream>>>(mask, Lis, Ris, scal, Aa);
    k_linear1<<<(NNODE * DHID) / 256, 256, 0, stream>>>(nf, W1, b1, x0, X, Xt);

    for (int it = 0; it < ITERS; ++it) {
        k_gemm<<<NNODE / 64, 128, 0, stream>>>(Aa, Xt, Y);
        k_update<<<(NNODE * DHID) / 256, 256, 0, stream>>>(x0, Y, X, Xt);
    }

    k_lineard<<<(NNODE * DOUT + 255) / 256, 256, 0, stream>>>(X, Wd, bd, xd);
    k_prepxdp<<<(NNODE * DPAD) / 256, 256, 0, stream>>>(xd, xdp);
    k_simloss<<<dim3(NNODE / 16, NNODE / 64), 32, 0, stream>>>(xdp, lossa);
    k_pool<<<(NNODE * DOUT + 255) / 256, 256, 0, stream>>>(xd, scal, pooled);
    k_final<<<1, 32, 0, stream>>>(pooled, lossa, W2, b2, outp);
}
